// GNN_15968688407182
// MI455X (gfx1250) — matrix operand order repaired and checked by reading, not compiled, not run on hardware
//
#include <hip/hip_runtime.h>
#include <stdint.h>

#define HF    128
#define FIN   64
#define EDIM  21

typedef __attribute__((ext_vector_type(16))) __bf16 v16bf;
typedef __attribute__((ext_vector_type(2)))  __bf16 bf16x2;
typedef __attribute__((ext_vector_type(8)))  float  v8f;

union FragAB { uint32_t u[8]; v16bf v; };

// pack two floats as a bf16 pair {lo=a, hi=b} via the native f32->bf16
// convert (lets clang emit the packed hardware cvt instead of shift/or ALU)
static __device__ __forceinline__ uint32_t pack2bf(float a, float b) {
  union { bf16x2 v; uint32_t u; } x;
  x.v = (bf16x2){(__bf16)a, (__bf16)b};
  return x.u;
}

// C = act(relu_a?(A)[MxK] @ W[KxN] + bias[N]); W row-major, row stride N.
// Requires K % 32 == 0 and N % 64 == 0 (true for every call here).
// 128x64 tile / 256-thread block; 8 waves (4x2), each wave = 32x32 output
// = 4 x v_wmma_f32_16x16x32_bf16 per 32-wide K step.
// Double-buffered LDS + register staging: one barrier per K step, global
// loads for step s+1 overlap the step-s WMMAs. Branch-free fills (clamped
// rows); interior-fast-path epilogue with pointer-walk addressing.
// RELUA applies ReLU to A during bf16 conversion (fuses the elementwise pass).
template <int RELU, int RELUA>
__global__ __launch_bounds__(256)
void gemm_bias_act(const float* __restrict__ A, const float* __restrict__ W,
                   const float* __restrict__ bias, float* __restrict__ C,
                   int M, int K, int N) {
  __shared__ unsigned short As[2][128][32];   // [buf][m][k] bf16
  __shared__ unsigned short Bs[2][64][32];    // [buf][n][k] bf16 (W transposed)
  const int tid  = threadIdx.x;
  const int lane = tid & 31, wid = tid >> 5;
  const int wrow = wid >> 1, wcol = wid & 1;
  const int m0 = blockIdx.y * 128, n0 = blockIdx.x * 64;
  const int hh = lane >> 4, la = lane & 15;

  v8f acc[4];
#pragma unroll
  for (int i = 0; i < 4; ++i) acc[i] = (v8f){0.f,0.f,0.f,0.f,0.f,0.f,0.f,0.f};

  float4 aReg[4];
  float  wReg[4][2];

  auto loadStep = [&](int kt) {      // global -> regs (branch-free, clamped)
#pragma unroll
    for (int i = 0; i < 4; ++i) {
      int idx = tid + 256 * i;
      int m = idx >> 3, q = idx & 7;
      int gm = m0 + m; gm = (gm < M) ? gm : (M - 1);
      aReg[i] = *(const float4*)(A + (long)gm * K + kt + q * 4);
    }
#pragma unroll
    for (int i = 0; i < 4; ++i) {
      int idx = tid + 256 * i;
      int n = idx & 63, kp = idx >> 6;              // kp = 0..15
      const float* wp = W + (long)(kt + 2 * kp) * N + n0 + n;
      wReg[i][0] = wp[0]; wReg[i][1] = wp[N];
    }
  };
  auto storeStep = [&](int buf) {    // regs -> LDS (packed bf16, fused ReLU-A)
#pragma unroll
    for (int i = 0; i < 4; ++i) {
      int idx = tid + 256 * i;
      int m = idx >> 3, q = idx & 7;
      float ax = aReg[i].x, ay = aReg[i].y, az = aReg[i].z, aw = aReg[i].w;
      if (RELUA) {
        ax = fmaxf(ax, 0.f); ay = fmaxf(ay, 0.f);
        az = fmaxf(az, 0.f); aw = fmaxf(aw, 0.f);
      }
      uint2 p; p.x = pack2bf(ax, ay); p.y = pack2bf(az, aw);
      *(uint2*)&As[buf][m][q * 4] = p;
    }
#pragma unroll
    for (int i = 0; i < 4; ++i) {
      int idx = tid + 256 * i;
      int n = idx & 63, kp = idx >> 6;
      *(uint32_t*)&Bs[buf][n][2 * kp] = pack2bf(wReg[i][0], wReg[i][1]);
    }
  };

  const int S = K >> 5;              // K steps (K % 32 == 0)
  loadStep(0);
  storeStep(0);
  __syncthreads();

  for (int s = 0; s < S; ++s) {
    if (s + 1 < S) loadStep((s + 1) << 5);     // overlap next tile with WMMAs
    const int cur = s & 1;
    FragAB a0, a1, b0, b1;
#pragma unroll
    for (int r = 0; r < 8; ++r) {
      // 16-bit A 16x32 layout: lanes<16: V0..3->K0..7, V4..7->K16..23; lanes>=16: +8
      int ka = ((r & 3) << 1) + ((r >= 4) ? 16 : 0) + hh * 8;
      a0.u[r] = *(const uint32_t*)&As[cur][wrow * 32 + la][ka];
      a1.u[r] = *(const uint32_t*)&As[cur][wrow * 32 + 16 + la][ka];
      // 16-bit B 32x16 layout (matches A): lanes<16: V0..3->K0..7, V4..7->K16..23; lanes>=16: +8
      int kb = ((r & 3) << 1) + ((r >= 4) ? 16 : 0) + hh * 8;
      b0.u[r] = *(const uint32_t*)&Bs[cur][wcol * 32 + la][kb];
      b1.u[r] = *(const uint32_t*)&Bs[cur][wcol * 32 + 16 + la][kb];
    }
    acc[0] = __builtin_amdgcn_wmma_f32_16x16x32_bf16(false, a0.v, false, b0.v,
                                                     (short)0, acc[0], false, false);
    acc[1] = __builtin_amdgcn_wmma_f32_16x16x32_bf16(false, a0.v, false, b1.v,
                                                     (short)0, acc[1], false, false);
    acc[2] = __builtin_amdgcn_wmma_f32_16x16x32_bf16(false, a1.v, false, b0.v,
                                                     (short)0, acc[2], false, false);
    acc[3] = __builtin_amdgcn_wmma_f32_16x16x32_bf16(false, a1.v, false, b1.v,
                                                     (short)0, acc[3], false, false);
    if (s + 1 < S) {
      storeStep((s + 1) & 1);        // write buffer last read at step s-1 (fenced)
      __syncthreads();
    }
  }

  // C/D layout: VGPR r -> (M = r + 8*(lane>=16), N = lane&15)
#pragma unroll
  for (int fi = 0; fi < 2; ++fi) {
#pragma unroll
    for (int ci = 0; ci < 2; ++ci) {
      const v8f  acv = acc[fi * 2 + ci];
      const int  n   = n0 + wcol * 32 + ci * 16 + la;
      const float bn = bias[n];
      const int  mb  = m0 + wrow * 32 + fi * 16 + hh * 8;
      float* p = C + (long)mb * N + n;
      if (mb + 7 < M) {                       // interior fast path: no branches
#pragma unroll
        for (int r = 0; r < 8; ++r) {
          float v = acv[r] + bn;
          if (RELU) v = fmaxf(v, 0.f);
          *p = v; p += N;
        }
      } else {
#pragma unroll
        for (int r = 0; r < 8; ++r) {
          if (mb + r < M) {
            float v = acv[r] + bn;
            if (RELU) v = fmaxf(v, 0.f);
            *p = v;
          }
          p += N;
        }
      }
    }
  }
}

// Fused ResGated edge stage: e = ea@We recomputed from LDS-resident We (K=21),
// gate = sigmoid(k[dst]+q[src]+2e), msg = gate*(v[src]+e), atomic scatter-add.
// One wave per edge; float4 per lane (coalesced 512B rows); next-edge prefetch.
__global__ __launch_bounds__(256)
void resgated_edge(const int* __restrict__ ei, const float* __restrict__ ea,
                   const float* __restrict__ We, const float* __restrict__ Kf,
                   const float* __restrict__ Qf, const float* __restrict__ Vf,
                   float* __restrict__ Agg, int E) {
  __shared__ float WeS[EDIM * HF];
  for (int i = threadIdx.x; i < EDIM * HF; i += 256) WeS[i] = We[i];
  __syncthreads();
  const int lane = threadIdx.x & 31, wid = threadIdx.x >> 5;
  const int base = blockIdx.x * 256;               // 256 edges per block
  const int f0   = lane * 4;                       // this lane's 4 features
#pragma unroll 1
  for (int it = 0; it < 32; ++it) {
    int e = base + it * 8 + wid;                   // one edge per wave
    if (e >= E) continue;
    int s = ei[e], d = ei[E + e];
    // prefetch next iteration's gathered rows (latency-bound random access)
    int en = e + 8;
    if (it < 31 && en < E) {
      int sn = ei[en], dn = ei[E + en];
      __builtin_prefetch(Kf + (long)dn * HF + f0, 0, 0);
      __builtin_prefetch(Qf + (long)sn * HF + f0, 0, 0);
      __builtin_prefetch(Vf + (long)sn * HF + f0, 0, 0);
    }
    float eav[EDIM];
#pragma unroll
    for (int j = 0; j < EDIM; ++j) eav[j] = ea[(long)e * EDIM + j];
    float4 ef = {0.f, 0.f, 0.f, 0.f};
#pragma unroll
    for (int j = 0; j < EDIM; ++j) {
      const float4 w = *(const float4*)&WeS[j * HF + f0];
      ef.x = fmaf(eav[j], w.x, ef.x);
      ef.y = fmaf(eav[j], w.y, ef.y);
      ef.z = fmaf(eav[j], w.z, ef.z);
      ef.w = fmaf(eav[j], w.w, ef.w);
    }
    const float4 kf = *(const float4*)(Kf + (long)d * HF + f0);
    const float4 qf = *(const float4*)(Qf + (long)s * HF + f0);
    const float4 vf = *(const float4*)(Vf + (long)s * HF + f0);
    float* ar = Agg + (long)d * HF + f0;
    {
      float g = kf.x + qf.x + 2.f * ef.x;
      atomicAdd(&ar[0], (vf.x + ef.x) / (1.f + __expf(-g)));
    }
    {
      float g = kf.y + qf.y + 2.f * ef.y;
      atomicAdd(&ar[1], (vf.y + ef.y) / (1.f + __expf(-g)));
    }
    {
      float g = kf.z + qf.z + 2.f * ef.z;
      atomicAdd(&ar[2], (vf.z + ef.z) / (1.f + __expf(-g)));
    }
    {
      float g = kf.w + qf.w + 2.f * ef.w;
      atomicAdd(&ar[3], (vf.w + ef.w) / (1.f + __expf(-g)));
    }
  }
}

__global__ void zero_f32(float* __restrict__ x, int n) {
  int i = blockIdx.x * blockDim.x + threadIdx.x;
  if (i < n) x[i] = 0.f;
}

__global__ __launch_bounds__(128)
void col_mean(const float* __restrict__ h, float* __restrict__ gmean,
              int M, int rowsPerBlock) {
  int col = threadIdx.x;
  int r0 = blockIdx.x * rowsPerBlock;
  int r1 = r0 + rowsPerBlock; if (r1 > M) r1 = M;
  float s = 0.f;
  for (int r = r0; r < r1; ++r) s += h[(long)r * HF + col];
  atomicAdd(&gmean[col], s * (1.f / (float)M));
}

// hb[n] = bl1[n] + sum_k gmean[k] * Wl1[128+k][n]  (folds the broadcast half)
__global__ __launch_bounds__(256)
void head_bias(const float* __restrict__ gmean, const float* __restrict__ Wl1,
               const float* __restrict__ bl1, float* __restrict__ hb) {
  int n = threadIdx.x;
  float acc = bl1[n];
  for (int k = 0; k < HF; ++k) acc = fmaf(gmean[k], Wl1[(long)(HF + k) * 256 + n], acc);
  hb[n] = acc;
}

__global__ __launch_bounds__(256)
void final_dot(const float* __restrict__ U, const float* __restrict__ Wl3,
               const float* __restrict__ bl3, float* __restrict__ out, int M) {
  int lane = threadIdx.x & 31, wid = threadIdx.x >> 5;
  int node = blockIdx.x * 8 + wid;
  if (node >= M) return;
  const float* u = U + (long)node * HF;
  float s = 0.f;
#pragma unroll
  for (int i = 0; i < 4; ++i) { int k = lane + 32 * i; s = fmaf(u[k], Wl3[k], s); }
#pragma unroll
  for (int off = 16; off > 0; off >>= 1) s += __shfl_xor(s, off, 32);
  if (lane == 0) out[node] = s + bl3[0];
}

extern "C" void kernel_launch(void* const* d_in, const int* in_sizes, int n_in,
                              void* d_out, int out_size, void* d_ws, size_t ws_size,
                              hipStream_t stream) {
  const float* G    = (const float*)d_in[0];
  const int*   ei   = (const int*)  d_in[1];
  const float* ea   = (const float*)d_in[2];
  const float *We1=(const float*)d_in[3],  *be1=(const float*)d_in[4];
  const float *We2=(const float*)d_in[5],  *be2=(const float*)d_in[6];
  const float *We3=(const float*)d_in[7],  *be3=(const float*)d_in[8];
  const float *c1Wk=(const float*)d_in[9],  *c1bk=(const float*)d_in[10];
  const float *c1Wq=(const float*)d_in[11], *c1bq=(const float*)d_in[12];
  const float *c1Wv=(const float*)d_in[13], *c1bv=(const float*)d_in[14];
  const float *c1We=(const float*)d_in[15];
  const float *c1Ws=(const float*)d_in[16], *c1bs=(const float*)d_in[17];
  const float *c2Wk=(const float*)d_in[18], *c2bk=(const float*)d_in[19];
  const float *c2Wq=(const float*)d_in[20], *c2bq=(const float*)d_in[21];
  const float *c2Wv=(const float*)d_in[22], *c2bv=(const float*)d_in[23];
  const float *c2We=(const float*)d_in[24];
  const float *c2Ws=(const float*)d_in[25], *c2bs=(const float*)d_in[26];
  const float *Wg1=(const float*)d_in[27], *bg1=(const float*)d_in[28];
  const float *Wg2=(const float*)d_in[29], *bg2=(const float*)d_in[30];
  const float *Wl1=(const float*)d_in[31], *bl1=(const float*)d_in[32];
  const float *Wl2=(const float*)d_in[33], *bl2=(const float*)d_in[34];
  const float *Wl3=(const float*)d_in[35], *bl3=(const float*)d_in[36];

  const int  Nn = in_sizes[0] / FIN;     // 50000
  const int  E  = in_sizes[1] / 2;       // 800000
  const long NH = (long)Nn * HF;

  float* ws     = (float*)d_ws;
  float* bufA   = ws;                    // h / u2
  float* bufB   = ws + 1 * NH;           // h tmp / conv2 agg / final h
  float* bufK   = ws + 2 * NH;           // k ; later U1 spans bufK..bufQ (Nx256)
  float* bufQ   = ws + 3 * NH;           // q
  float* bufV   = ws + 4 * NH;           // v
  float* bufAgg = ws + 5 * NH;           // conv1 agg (skip-preloaded)
  float* gmean  = ws + 6 * NH;
  float* hb     = gmean + HF;            // 256 fused head biases

  const dim3 blk(256);
  auto gg = [](int M, int N) { return dim3((unsigned)((N + 63) / 64),
                                           (unsigned)((M + 127) / 128)); };
  const int eBlocks = (E + 255) / 256;

  // node embedding MLP
  gemm_bias_act<1,0><<<gg(Nn, HF), blk, 0, stream>>>(G,    We1, be1, bufA, Nn, FIN, HF);
  gemm_bias_act<1,0><<<gg(Nn, HF), blk, 0, stream>>>(bufA, We2, be2, bufB, Nn, HF,  HF);
  gemm_bias_act<1,0><<<gg(Nn, HF), blk, 0, stream>>>(bufB, We3, be3, bufA, Nn, HF,  HF);

  // conv1: k,q,v + skip (skip written directly into agg buffer)
  gemm_bias_act<0,0><<<gg(Nn, HF), blk, 0, stream>>>(bufA, c1Wk, c1bk, bufK,   Nn, HF, HF);
  gemm_bias_act<0,0><<<gg(Nn, HF), blk, 0, stream>>>(bufA, c1Wq, c1bq, bufQ,   Nn, HF, HF);
  gemm_bias_act<0,0><<<gg(Nn, HF), blk, 0, stream>>>(bufA, c1Wv, c1bv, bufV,   Nn, HF, HF);
  gemm_bias_act<0,0><<<gg(Nn, HF), blk, 0, stream>>>(bufA, c1Ws, c1bs, bufAgg, Nn, HF, HF);
  resgated_edge<<<eBlocks, blk, 0, stream>>>(ei, ea, c1We, bufK, bufQ, bufV, bufAgg, E);

  // conv2: x = relu(bufAgg), fused into the A-side of each GEMM
  gemm_bias_act<0,1><<<gg(Nn, HF), blk, 0, stream>>>(bufAgg, c2Wk, c2bk, bufK, Nn, HF, HF);
  gemm_bias_act<0,1><<<gg(Nn, HF), blk, 0, stream>>>(bufAgg, c2Wq, c2bq, bufQ, Nn, HF, HF);
  gemm_bias_act<0,1><<<gg(Nn, HF), blk, 0, stream>>>(bufAgg, c2Wv, c2bv, bufV, Nn, HF, HF);
  gemm_bias_act<0,1><<<gg(Nn, HF), blk, 0, stream>>>(bufAgg, c2Ws, c2bs, bufB, Nn, HF, HF);
  resgated_edge<<<eBlocks, blk, 0, stream>>>(ei, ea, c2We, bufK, bufQ, bufV, bufB, E);

  // graph-level MLP (ReLU of conv2 output fused into Wg1's A-side)
  gemm_bias_act<1,1><<<gg(Nn, HF), blk, 0, stream>>>(bufB, Wg1, bg1, bufA, Nn, HF, HF);
  gemm_bias_act<0,0><<<gg(Nn, HF), blk, 0, stream>>>(bufA, Wg2, bg2, bufB, Nn, HF, HF);

  // graph mean -> fused head bias (replaces the [N,2H] concat)
  zero_f32<<<1, 128, 0, stream>>>(gmean, HF);
  int rpb = (Nn + 199) / 200;
  col_mean<<<200, 128, 0, stream>>>(bufB, gmean, Nn, rpb);
  head_bias<<<1, 256, 0, stream>>>(gmean, Wl1, bl1, hb);

  // head: u1 = relu(h @ Wl1[0:128,:] + hb)  (Wl1 row stride 256; top half read)
  gemm_bias_act<1,0><<<gg(Nn, 256), blk, 0, stream>>>(bufB, Wl1, hb, bufK, Nn, HF, 256);
  gemm_bias_act<1,0><<<gg(Nn, HF),  blk, 0, stream>>>(bufK, Wl2, bl2, bufA, Nn, 256, HF);
  final_dot<<<(Nn + 7) / 8, blk, 0, stream>>>(bufA, Wl3, bl3, (float*)d_out, Nn);
}
